// HippoLegSCell_41489384079632
// MI455X (gfx1250) — matrix operand order repaired and checked by reading, not compiled, not run on hardware
//
#include <hip/hip_runtime.h>

#define N_STATE 256
#define SEQLEN  1024
#define NBATCH  64

typedef __attribute__((ext_vector_type(16))) __bf16 v16bf;
typedef __attribute__((ext_vector_type(8)))  float  v8f;

// ---- float -> bf16 helpers -------------------------------------------------
static __device__ __forceinline__ unsigned short f2bf(float x) {
  unsigned int u = __float_as_uint(x);
  unsigned int r = u + 0x7FFFu + ((u >> 16) & 1u);
  return (unsigned short)(r >> 16);
}
static __device__ __forceinline__ unsigned int pkbf(float a, float b) {
#if __has_builtin(__builtin_amdgcn_cvt_pk_bf16_f32)
  typedef __attribute__((ext_vector_type(2))) __bf16 v2bf;
  union { v2bf v; unsigned int u; } x;
  x.v = __builtin_amdgcn_cvt_pk_bf16_f32(a, b);
  return x.u;
#else
  return (unsigned int)f2bf(a) | ((unsigned int)f2bf(b) << 16);
#endif
}
// Assemble a v16bf WMMA operand from two 16-byte LDS chunks.
static __device__ __forceinline__ v16bf ld_frag(const unsigned short* p0,
                                                const unsigned short* p1) {
  union { uint4 u[2]; v16bf v; } x;
  x.u[0] = *(const uint4*)p0;
  x.u[1] = *(const uint4*)p1;
  return x.v;
}

// ---- CDNA5 async global->LDS copy (ASYNCcnt-tracked) -----------------------
// INST_OFFSET is added to both the global and LDS addresses, so one address
// pair per thread serves all 16 chunks of its tile slice.
#define ASYNC_LD16(loff, gptr, imm)                                        \
  asm volatile("global_load_async_to_lds_b128 %0, %1, off offset:%2"       \
               :: "v"(loff), "v"(gptr), "n"(imm) : "memory")
#define WAIT_ASYNCCNT(n) \
  asm volatile("s_wait_asynccnt %0" :: "n"(n) : "memory")

// Issue one 128KB bf16 tile (65536 ushorts) as 16 async b128 ops per thread.
static __device__ __forceinline__ void issue_tile_async(
    const unsigned short* gsrc, const unsigned short* ldst, int tid) {
  unsigned loff = (unsigned)(uintptr_t)ldst + (unsigned)tid * 16u;
  const char* g = (const char*)gsrc + (size_t)tid * 16u;
  ASYNC_LD16(loff, g, 0);
  ASYNC_LD16(loff, g, 8192);
  ASYNC_LD16(loff, g, 16384);
  ASYNC_LD16(loff, g, 24576);
  ASYNC_LD16(loff, g, 32768);
  ASYNC_LD16(loff, g, 40960);
  ASYNC_LD16(loff, g, 49152);
  ASYNC_LD16(loff, g, 57344);
  ASYNC_LD16(loff, g, 65536);
  ASYNC_LD16(loff, g, 73728);
  ASYNC_LD16(loff, g, 81920);
  ASYNC_LD16(loff, g, 90112);
  ASYNC_LD16(loff, g, 98304);
  ASYNC_LD16(loff, g, 106496);
  ASYNC_LD16(loff, g, 114688);
  ASYNC_LD16(loff, g, 122880);
}

// ---------------------------------------------------------------------------
// Prepass: A_stack f32 -> bf16 into workspace (one-time, fully parallel).
// ---------------------------------------------------------------------------
__global__ __launch_bounds__(256) void convert_bf16_kernel(
    const float4* __restrict__ src, uint2* __restrict__ dst, long n4) {
  long i = (long)blockIdx.x * 256 + threadIdx.x;
  const long stride = (long)gridDim.x * 256;
  for (; i < n4; i += stride) {
    float4 v = src[i];
    dst[i] = make_uint2(pkbf(v.x, v.y), pkbf(v.z, v.w));
  }
}

// ---------------------------------------------------------------------------
// Scan (fast path): A_stack pre-converted to bf16 in ws. 4 sync-free
// workgroups (batch split, M=16). Per step, tile t+1 streams into the
// alternate LDS buffer via GLOBAL_LOAD_ASYNC_TO_LDS_B128 while 16 waves
// consume tile t with 8x v_wmma_f32_16x16x32_bf16 each.
// LDS: 2 x 128KB A buffers + 2 x 8KB bf16 state = 272KB of 320KB.
// ---------------------------------------------------------------------------
__global__ __launch_bounds__(512) void hippo_scan_async_kernel(
    const float* __restrict__ inputs,           // (64, 1024)
    const unsigned short* __restrict__ Abf,     // (1024, 256, 256) bf16
    const float* __restrict__ B_stack,          // (1024, 256)
    float* __restrict__ cs)                     // (64, 1024, 256)
{
  extern __shared__ unsigned short lds[];
  const int TILE_E = N_STATE * N_STATE;               // ushorts per tile
  unsigned short* Ab0 = lds;
  unsigned short* Ab1 = lds + TILE_E;
  unsigned short* Cb0 = lds + 2 * TILE_E;
  unsigned short* Cb1 = Cb0 + 16 * N_STATE;

  const int tid  = threadIdx.x;
  const int lane = tid & 31;
  const int wave = tid >> 5;          // 0..15 -> n-tile
  const int half = lane >> 4;
  const int l16  = lane & 15;
  const int n0   = wave * 16;
  const int bg16 = blockIdx.x * 16;

  // zero both state buffers (c0 = 0)
  {
    unsigned int* z = (unsigned int*)Cb0;
    for (int i = tid; i < 16 * N_STATE; i += 512) z[i] = 0u;
  }

  // prime the pipeline: tile 0 -> Ab0
  issue_tile_async(Abf, Ab0, tid);

  for (int t = 0; t < SEQLEN; ++t) {
    __syncthreads();  // step t-1 done reading Abuf[(t+1)&1] and C buffers

    if (t + 1 < SEQLEN) {
      issue_tile_async(Abf + (size_t)(t + 1) * TILE_E,
                       ((t + 1) & 1) ? Ab1 : Ab0, tid);
      WAIT_ASYNCCNT(16);  // in-order completion => tile t has landed
    } else {
      WAIT_ASYNCCNT(0);
    }
    __syncthreads();  // cross-wave visibility of tile t

    const unsigned short* At   = (t & 1) ? Ab1 : Ab0;
    const unsigned short* Cin  = (t & 1) ? Cb1 : Cb0;
    unsigned short*       Cout = (t & 1) ? Cb0 : Cb1;

    v8f acc = {0.f, 0.f, 0.f, 0.f, 0.f, 0.f, 0.f, 0.f};
#pragma unroll
    for (int kc = 0; kc < 8; ++kc) {
      const int k0 = kc * 32;
      v16bf a = ld_frag(Cin + l16 * N_STATE + k0 + half * 8,
                        Cin + l16 * N_STATE + k0 + half * 8 + 16);
      const unsigned short* bp = At + (n0 + l16) * N_STATE + k0 + half * 8;
      v16bf b = ld_frag(bp, bp + 16);
      acc = __builtin_amdgcn_wmma_f32_16x16x32_bf16(false, a, false, b,
                                                    (short)0, acc, false, false);
    }

    const int n = n0 + l16;
    const float bt = B_stack[(size_t)t * N_STATE + n];
#pragma unroll
    for (int r = 0; r < 8; ++r) {
      const int bl = half * 8 + r;
      const float f = inputs[(size_t)(bg16 + bl) * SEQLEN + t];
      const float v = acc[r] + bt * f;
      cs[(((size_t)(bg16 + bl)) * SEQLEN + t) * N_STATE + n] = v;
      Cout[bl * N_STATE + n] = f2bf(v);
    }
  }
}

// ---------------------------------------------------------------------------
// Scan (fallback, no workspace): in-loop f32->bf16 conversion of A_t.
// ---------------------------------------------------------------------------
__global__ __launch_bounds__(512) void hippo_scan_kernel(
    const float* __restrict__ inputs, const float* __restrict__ A_stack,
    const float* __restrict__ B_stack, float* __restrict__ cs)
{
  extern __shared__ unsigned short lds[];
  unsigned short* Atile = lds;
  unsigned short* Cbuf0 = lds + N_STATE * N_STATE;
  unsigned short* Cbuf1 = Cbuf0 + 16 * N_STATE;

  const int tid  = threadIdx.x;
  const int lane = tid & 31;
  const int wave = tid >> 5;
  const int half = lane >> 4;
  const int l16  = lane & 15;
  const int n0   = wave * 16;
  const int bg16 = blockIdx.x * 16;

  {
    unsigned int* z = (unsigned int*)Cbuf0;
    for (int i = tid; i < 16 * N_STATE; i += 512) z[i] = 0u;
  }

  for (int t = 0; t < SEQLEN; ++t) {
    __syncthreads();
    const float4* src = (const float4*)(A_stack + (size_t)t * (N_STATE * N_STATE));
    uint2* dst = (uint2*)Atile;
#pragma unroll 4
    for (int c = tid; c < (N_STATE * N_STATE) / 4; c += 512) {
      float4 v = src[c];
      dst[c] = make_uint2(pkbf(v.x, v.y), pkbf(v.z, v.w));
    }
    __syncthreads();

    if (t + 1 < SEQLEN) {
      const char* np =
          (const char*)(A_stack + (size_t)(t + 1) * (N_STATE * N_STATE)) + tid * 512;
      __builtin_prefetch(np, 0, 0);
      __builtin_prefetch(np + 256, 0, 0);
    }

    const unsigned short* Cin  = (t & 1) ? Cbuf1 : Cbuf0;
    unsigned short*       Cout = (t & 1) ? Cbuf0 : Cbuf1;

    v8f acc = {0.f, 0.f, 0.f, 0.f, 0.f, 0.f, 0.f, 0.f};
#pragma unroll
    for (int kc = 0; kc < 8; ++kc) {
      const int k0 = kc * 32;
      v16bf a = ld_frag(Cin + l16 * N_STATE + k0 + half * 8,
                        Cin + l16 * N_STATE + k0 + half * 8 + 16);
      const unsigned short* bp = Atile + (n0 + l16) * N_STATE + k0 + half * 8;
      v16bf b = ld_frag(bp, bp + 16);
      acc = __builtin_amdgcn_wmma_f32_16x16x32_bf16(false, a, false, b,
                                                    (short)0, acc, false, false);
    }

    const int n = n0 + l16;
    const float bt = B_stack[(size_t)t * N_STATE + n];
#pragma unroll
    for (int r = 0; r < 8; ++r) {
      const int bl = half * 8 + r;
      const float f = inputs[(size_t)(bg16 + bl) * SEQLEN + t];
      const float v = acc[r] + bt * f;
      cs[(((size_t)(bg16 + bl)) * SEQLEN + t) * N_STATE + n] = v;
      Cout[bl * N_STATE + n] = f2bf(v);
    }
  }
}

// ---------------------------------------------------------------------------
// recon[b,l,m] = sum_n cs[b,l,n] * eval_mat[m,n]; 64x64 tile per block,
// K=256 staged in 64KB LDS as bf16, 8 waves x 16 WMMAs.
// ---------------------------------------------------------------------------
__global__ __launch_bounds__(256) void hippo_recon_kernel(
    const float* __restrict__ cs, const float* __restrict__ eval_mat,
    float* __restrict__ recon)
{
  extern __shared__ unsigned short lds[];
  unsigned short* Ct = lds;
  unsigned short* Et = lds + 64 * N_STATE;

  const int tid  = threadIdx.x;
  const int lane = tid & 31;
  const int wave = tid >> 5;
  const int half = lane >> 4;
  const int l16  = lane & 15;
  const int mblk = blockIdx.x * 64;
  const int lblk = blockIdx.y * 64;
  const int b    = blockIdx.z;
  const int lw   = wave >> 1;
  const int mw   = wave & 1;

  {
    const float4* s0 = (const float4*)(cs + (((size_t)b * SEQLEN) + lblk) * N_STATE);
    const float4* s1 = (const float4*)(eval_mat + (size_t)mblk * N_STATE);
    uint2* d0 = (uint2*)Ct;
    uint2* d1 = (uint2*)Et;
#pragma unroll 4
    for (int c = tid; c < (64 * N_STATE) / 4; c += 256) {
      float4 v = s0[c];
      d0[c] = make_uint2(pkbf(v.x, v.y), pkbf(v.z, v.w));
      float4 w = s1[c];
      d1[c] = make_uint2(pkbf(w.x, w.y), pkbf(w.z, w.w));
    }
  }
  __syncthreads();

  v8f acc0 = {0.f, 0.f, 0.f, 0.f, 0.f, 0.f, 0.f, 0.f};
  v8f acc1 = {0.f, 0.f, 0.f, 0.f, 0.f, 0.f, 0.f, 0.f};
#pragma unroll
  for (int kc = 0; kc < 8; ++kc) {
    const int k0 = kc * 32;
    v16bf a = ld_frag(Ct + (lw * 16 + l16) * N_STATE + k0 + half * 8,
                      Ct + (lw * 16 + l16) * N_STATE + k0 + half * 8 + 16);
    const unsigned short* bp0 = Et + (mw * 32 + l16) * N_STATE + k0 + half * 8;
    const unsigned short* bp1 = Et + (mw * 32 + 16 + l16) * N_STATE + k0 + half * 8;
    v16bf b0 = ld_frag(bp0, bp0 + 16);
    v16bf b1 = ld_frag(bp1, bp1 + 16);
    acc0 = __builtin_amdgcn_wmma_f32_16x16x32_bf16(false, a, false, b0,
                                                   (short)0, acc0, false, false);
    acc1 = __builtin_amdgcn_wmma_f32_16x16x32_bf16(false, a, false, b1,
                                                   (short)0, acc1, false, false);
  }

  const size_t base =
      ((size_t)b * SEQLEN + (size_t)(lblk + lw * 16 + half * 8)) * SEQLEN +
      (size_t)(mblk + mw * 32 + l16);
#pragma unroll
  for (int r = 0; r < 8; ++r) {
    recon[base + (size_t)r * SEQLEN]      = acc0[r];
    recon[base + (size_t)r * SEQLEN + 16] = acc1[r];
  }
}

extern "C" void kernel_launch(void* const* d_in, const int* in_sizes, int n_in,
                              void* d_out, int out_size, void* d_ws, size_t ws_size,
                              hipStream_t stream) {
  (void)in_sizes; (void)n_in; (void)out_size;
  const float* inputs   = (const float*)d_in[0];
  const float* A_stack  = (const float*)d_in[1];
  const float* B_stack  = (const float*)d_in[2];
  const float* eval_mat = (const float*)d_in[3];

  float* cs    = (float*)d_out;
  float* recon = cs + (size_t)NBATCH * SEQLEN * N_STATE;

  const size_t a_elems   = (size_t)SEQLEN * N_STATE * N_STATE;
  const size_t ws_needed = a_elems * sizeof(unsigned short);

  if (ws_size >= ws_needed) {
    // Fast path: bf16 prepass + async double-buffered scan.
    convert_bf16_kernel<<<dim3(8192), dim3(256), 0, stream>>>(
        (const float4*)A_stack, (uint2*)d_ws, (long)(a_elems / 4));
    size_t lds_scan =
        (2 * (size_t)N_STATE * N_STATE + 2 * 16 * N_STATE) * sizeof(unsigned short);
    hippo_scan_async_kernel<<<dim3(4), dim3(512), lds_scan, stream>>>(
        inputs, (const unsigned short*)d_ws, B_stack, cs);
  } else {
    // Fallback: convert A_t in-loop.
    size_t lds_scan =
        ((size_t)N_STATE * N_STATE + 2 * 16 * N_STATE) * sizeof(unsigned short);
    hippo_scan_kernel<<<dim3(4), dim3(512), lds_scan, stream>>>(
        inputs, A_stack, B_stack, cs);
  }

  size_t lds_recon = (size_t)(2 * 64 * N_STATE) * sizeof(unsigned short);
  hippo_recon_kernel<<<dim3(16, 16, 64), dim3(256), lds_recon, stream>>>(
      cs, eval_mat, recon);
}
